// MHGAT_29712583754282
// MI455X (gfx1250) — compile-verified, matrix operand order checked by reading, not run on hardware
//
#include <hip/hip_runtime.h>
#include <stdint.h>

#define N_NODES 20000
#define N_EDGES 320000
#define E_TOT   (N_EDGES + N_NODES)
#define V_DIM   768
#define T_DIM   384
#define IN_DIM  (V_DIM + T_DIM)
#define HID     128
#define HEADS   4
#define LAYERS  2
#define NEG_SLOPE 0.2f

typedef __attribute__((ext_vector_type(16))) __bf16    v16bf;
typedef __attribute__((ext_vector_type(8)))  float     v8f;
typedef __attribute__((ext_vector_type(4)))  unsigned  u32x4;

union FragBF { u32x4 q[2]; v16bf v; };

// ---------- helpers ----------
__device__ __forceinline__ unsigned short f2bf_bits(float f) {
  unsigned u = __float_as_uint(f);
  unsigned r = u + 0x7FFFu + ((u >> 16) & 1u);   // round-to-nearest-even
  return (unsigned short)(r >> 16);
}
// order-preserving float -> uint key for atomicMax
__device__ __forceinline__ unsigned fkey(float f) {
  unsigned u = __float_as_uint(f);
  return (u & 0x80000000u) ? ~u : (u | 0x80000000u);
}
__device__ __forceinline__ float funkey(unsigned k) {
  unsigned u = (k & 0x80000000u) ? (k & 0x7FFFFFFFu) : ~k;
  return __uint_as_float(u);
}

// ---------- conversion kernels ----------
__global__ void k_f32_to_bf16(const float* __restrict__ in,
                              unsigned short* __restrict__ out, long n) {
  long i = (long)blockIdx.x * blockDim.x + threadIdx.x;
  if (i < n) out[i] = f2bf_bits(in[i]);
}

// W [K,N] fp32 -> Wt [N,K] bf16
__global__ void k_transpose_bf16(const float* __restrict__ W,
                                 unsigned short* __restrict__ Wt, int K, int N) {
  long i = (long)blockIdx.x * blockDim.x + threadIdx.x;
  if (i >= (long)K * N) return;
  int k = (int)(i / N), nn = (int)(i % N);
  Wt[(size_t)nn * K + k] = f2bf_bits(W[i]);
}

__global__ void k_fill_f32(float* p, float v, long n) {
  long i = (long)blockIdx.x * blockDim.x + threadIdx.x;
  if (i < n) p[i] = v;
}
__global__ void k_fill_u32(unsigned* p, unsigned v, long n) {
  long i = (long)blockIdx.x * blockDim.x + threadIdx.x;
  if (i < n) p[i] = v;
}

// ---------- WMMA bf16 GEMM, 2x2 register blocking ----------
// C[M,N] = act(A[M,lda(K used)] @ Bt[N,K]^T + bias)
// Each wave computes a 32x32 patch: 4 WMMAs per k-chunk off 2 A-frags + 2 B-frags.
// grid.x = M/32; (grid.y * nwaves) * 32 covers N. Wave-uniform guards keep EXEC full.
__global__ void k_gemm_wmma(const unsigned short* __restrict__ A, int lda,
                            const unsigned short* __restrict__ Bt,
                            const float* __restrict__ bias,
                            float* __restrict__ C, int ldc,
                            int N, int K, int act) {
  const int wave = threadIdx.x >> 5;
  const int lane = threadIdx.x & 31;
  const int nw   = blockDim.x >> 5;
  const int m0   = blockIdx.x * 32;
  const int n0   = (blockIdx.y * nw + wave) * 32;
  if (n0 >= N) return;                          // wave-uniform

  const int khalf = lane >> 4;                  // 0: K {0..7,16..23}, 1: {8..15,24..31}
  const unsigned short* arow0 = A  + (size_t)(m0 + (lane & 15)) * lda;
  const unsigned short* arow1 = arow0 + (size_t)16 * lda;
  const unsigned short* brow0 = Bt + (size_t)(n0 + (lane & 15)) * K;
  const unsigned short* brow1 = brow0 + (size_t)16 * K;

  v8f acc00 = {0.f,0.f,0.f,0.f,0.f,0.f,0.f,0.f};
  v8f acc01 = acc00, acc10 = acc00, acc11 = acc00;

  for (int k0 = 0; k0 < K; k0 += 32) {
    const int o0 = k0 + khalf * 8;
    const int o1 = k0 + 16 + khalf * 8;
    // prefetch next k-chunk (lowers to global_prefetch_b8; speculative, OOB-safe)
    __builtin_prefetch(arow0 + k0 + 32, 0, 0);
    __builtin_prefetch(brow0 + k0 + 32, 0, 0);

    FragBF a0, a1, b0, b1;
    a0.q[0] = *(const u32x4*)(arow0 + o0);  a0.q[1] = *(const u32x4*)(arow0 + o1);
    a1.q[0] = *(const u32x4*)(arow1 + o0);  a1.q[1] = *(const u32x4*)(arow1 + o1);
    b0.q[0] = *(const u32x4*)(brow0 + o0);  b0.q[1] = *(const u32x4*)(brow0 + o1);
    b1.q[0] = *(const u32x4*)(brow1 + o0);  b1.q[1] = *(const u32x4*)(brow1 + o1);

    acc00 = __builtin_amdgcn_wmma_f32_16x16x32_bf16(false, a0.v, false, b0.v,
                                                    (short)0, acc00, false, false);
    acc01 = __builtin_amdgcn_wmma_f32_16x16x32_bf16(false, a0.v, false, b1.v,
                                                    (short)0, acc01, false, false);
    acc10 = __builtin_amdgcn_wmma_f32_16x16x32_bf16(false, a1.v, false, b0.v,
                                                    (short)0, acc10, false, false);
    acc11 = __builtin_amdgcn_wmma_f32_16x16x32_bf16(false, a1.v, false, b1.v,
                                                    (short)0, acc11, false, false);
  }

  const int col0 = n0 + (lane & 15);
  const int row0 = m0 + (lane >> 4) * 8;        // VGPR r -> row r (+8 for lanes 16-31)
  const float bv0 = bias ? bias[col0]      : 0.0f;
  const float bv1 = bias ? bias[col0 + 16] : 0.0f;
#pragma unroll
  for (int r = 0; r < 8; ++r) {
    float v00 = acc00[r] + bv0, v01 = acc01[r] + bv1;
    float v10 = acc10[r] + bv0, v11 = acc11[r] + bv1;
    if (act == 1) {
      v00 = fmaxf(v00, 0.f); v01 = fmaxf(v01, 0.f);
      v10 = fmaxf(v10, 0.f); v11 = fmaxf(v11, 0.f);
    }
    C[(size_t)(row0 + r)      * ldc + col0]      = v00;
    C[(size_t)(row0 + r)      * ldc + col0 + 16] = v01;
    C[(size_t)(row0 + 16 + r) * ldc + col0]      = v10;
    C[(size_t)(row0 + 16 + r) * ldc + col0 + 16] = v11;
  }
}

// ---------- attention logits: al[n,h] = <xh[n,h,:], a[h,:]> ----------
__global__ void k_attn_logits(const float* __restrict__ xh,
                              const float* __restrict__ a_src,
                              const float* __restrict__ a_dst,
                              float* __restrict__ al_src,
                              float* __restrict__ al_dst) {
  const int node = blockIdx.x;
  const int h = threadIdx.x >> 5, lane = threadIdx.x & 31;
  const float* v = xh + (size_t)node * HEADS * HID + h * HID;
  float ps = 0.f, pd = 0.f;
  for (int c = lane; c < HID; c += 32) {
    float xv = v[c];
    ps += xv * a_src[h * HID + c];
    pd += xv * a_dst[h * HID + c];
  }
  for (int off = 16; off > 0; off >>= 1) {
    ps += __shfl_xor(ps, off, 32);
    pd += __shfl_xor(pd, off, 32);
  }
  if (lane == 0) {
    al_src[node * HEADS + h] = ps;
    al_dst[node * HEADS + h] = pd;
  }
}

__device__ __forceinline__ void edge_sd(const int* __restrict__ src,
                                        const int* __restrict__ dst,
                                        int e, int& s, int& d) {
  if (e < N_EDGES) { s = src[e]; d = dst[e]; }
  else             { s = d = e - N_EDGES; }   // self-loops appended
}

// ---------- pass 1: segment max of leaky_relu logits ----------
__global__ void k_edge_max(const int* __restrict__ src, const int* __restrict__ dst,
                           const float* __restrict__ al_src, const float* __restrict__ al_dst,
                           unsigned* __restrict__ mkey) {
  long t = (long)blockIdx.x * blockDim.x + threadIdx.x;
  if (t >= (long)E_TOT * HEADS) return;
  int e = (int)(t / HEADS), h = (int)(t % HEADS), s, d;
  edge_sd(src, dst, e, s, d);
  float v = al_src[s * HEADS + h] + al_dst[d * HEADS + h];
  v = v > 0.f ? v : NEG_SLOPE * v;
  atomicMax(&mkey[d * HEADS + h], fkey(v));
}

// ---------- pass 2: exp(e - max), segment sum ----------
__global__ void k_edge_expsum(const int* __restrict__ src, const int* __restrict__ dst,
                              const float* __restrict__ al_src, const float* __restrict__ al_dst,
                              const unsigned* __restrict__ mkey,
                              float* __restrict__ ebuf, float* __restrict__ ssum) {
  long t = (long)blockIdx.x * blockDim.x + threadIdx.x;
  if (t >= (long)E_TOT * HEADS) return;
  int e = (int)(t / HEADS), h = (int)(t % HEADS), s, d;
  edge_sd(src, dst, e, s, d);
  float v = al_src[s * HEADS + h] + al_dst[d * HEADS + h];
  v = v > 0.f ? v : NEG_SLOPE * v;
  float m = funkey(mkey[d * HEADS + h]);
  float ex = __expf(v - m);
  ebuf[t] = ex;
  atomicAdd(&ssum[d * HEADS + h], ex);
}

// ---------- pass 3: agg[dst,h,:] += alpha * xh[src,h,:] ----------
// one block per edge; 4 waves = 4 heads; lane covers 4 channels
__global__ void k_edge_aggregate(const int* __restrict__ src, const int* __restrict__ dst,
                                 const float* __restrict__ ebuf, const float* __restrict__ ssum,
                                 const float* __restrict__ xh, float* __restrict__ agg) {
  int e = blockIdx.x;
  int h = threadIdx.x >> 5, lane = threadIdx.x & 31, s, d;
  edge_sd(src, dst, e, s, d);
  float alpha = ebuf[(size_t)e * HEADS + h] / (ssum[d * HEADS + h] + 1e-16f);
  const float* xrow = xh + (size_t)s * HEADS * HID + h * HID;
  float* orow = agg + (size_t)d * HEADS * HID + h * HID;
  for (int c = lane; c < HID; c += 32)
    atomicAdd(&orow[c], xrow[c] * alpha);
}

// ---------- head-mean + bias + ELU + residual ----------
__global__ void k_node_update(const float* __restrict__ agg, const float* __restrict__ bias,
                              float* __restrict__ h_io) {
  long i = (long)blockIdx.x * blockDim.x + threadIdx.x;
  if (i >= (long)N_NODES * HID) return;
  int node = (int)(i / HID), c = (int)(i % HID);
  float m = 0.f;
#pragma unroll
  for (int h = 0; h < HEADS; ++h)
    m += agg[(size_t)node * HEADS * HID + h * HID + c];
  m = m * (1.0f / HEADS) + bias[c];
  float e = m > 0.f ? m : (__expf(m) - 1.0f);   // ELU
  h_io[i] = e + h_io[i];
}

// ---------- sigmoid modal gate + blended embedding ----------
__global__ void k_modal_final(const float* __restrict__ hv, const float* __restrict__ ht,
                              const float* __restrict__ mw, const float* __restrict__ mb,
                              float* __restrict__ out) {
  int node = blockIdx.x * (blockDim.x >> 5) + (threadIdx.x >> 5);
  int lane = threadIdx.x & 31;
  if (node >= N_NODES) return;
  const float* pv = hv + (size_t)node * HID;
  const float* pt = ht + (size_t)node * HID;
  float p = 0.f;
  for (int c = lane; c < HID; c += 32)
    p += pv[c] * mw[c] + pt[c] * mw[HID + c];
  for (int off = 16; off > 0; off >>= 1) p += __shfl_xor(p, off, 32);
  float beta = 1.0f / (1.0f + __expf(-(p + mb[0])));
  for (int c = lane; c < HID; c += 32)
    out[(size_t)node * HID + c] = beta * pv[c] + (1.0f - beta) * pt[c];
}

// =================== host orchestration ===================
extern "C" void kernel_launch(void* const* d_in, const int* in_sizes, int n_in,
                              void* d_out, int out_size, void* d_ws, size_t ws_size,
                              hipStream_t stream) {
  const float* x          = (const float*)d_in[0];
  const int*   ei         = (const int*)d_in[1];
  const float* vis_proj_w = (const float*)d_in[2];
  const float* vis_proj_b = (const float*)d_in[3];
  const float* txt_proj_w = (const float*)d_in[4];
  const float* txt_proj_b = (const float*)d_in[5];
  const float* vis_gat_w  = (const float*)d_in[6];
  const float* vis_gat_as = (const float*)d_in[7];
  const float* vis_gat_ad = (const float*)d_in[8];
  const float* vis_gat_b  = (const float*)d_in[9];
  const float* txt_gat_w  = (const float*)d_in[10];
  const float* txt_gat_as = (const float*)d_in[11];
  const float* txt_gat_ad = (const float*)d_in[12];
  const float* txt_gat_b  = (const float*)d_in[13];
  const float* modal_w    = (const float*)d_in[14];
  const float* modal_b    = (const float*)d_in[15];
  const float* vhead_w    = (const float*)d_in[16];
  const float* vhead_b    = (const float*)d_in[17];
  const float* thead_w    = (const float*)d_in[18];
  const float* thead_b    = (const float*)d_in[19];

  const int* src = ei;
  const int* dst = ei + N_EDGES;

  // ---- carve workspace ----
  char* base = (char*)d_ws;
  size_t off = 0;
  auto carve = [&](size_t bytes) -> void* {
    void* p = base + off;
    off += (bytes + 255) & ~(size_t)255;
    return p;
  };
  unsigned short* x_bf     = (unsigned short*)carve((size_t)N_NODES * IN_DIM * 2);
  unsigned short* wt_vproj = (unsigned short*)carve((size_t)HID * V_DIM * 2);
  unsigned short* wt_tproj = (unsigned short*)carve((size_t)HID * T_DIM * 2);
  unsigned short* wt_gat[4];
  for (int i = 0; i < 4; ++i)   // [vL0, vL1, tL0, tL1], each (HEADS*HID) x HID transposed
    wt_gat[i] = (unsigned short*)carve((size_t)HEADS * HID * HID * 2);
  unsigned short* wt_vhead = (unsigned short*)carve((size_t)HID * HID * 2);
  unsigned short* wt_thead = (unsigned short*)carve((size_t)HID * HID * 2);
  float* h_v  = (float*)carve((size_t)N_NODES * HID * 4);
  float* h_t  = (float*)carve((size_t)N_NODES * HID * 4);
  unsigned short* h_bf = (unsigned short*)carve((size_t)N_NODES * HID * 2);
  float* xh   = (float*)carve((size_t)N_NODES * HEADS * HID * 4);
  float* agg  = (float*)carve((size_t)N_NODES * HEADS * HID * 4);
  float* al_s = (float*)carve((size_t)N_NODES * HEADS * 4);
  float* al_d = (float*)carve((size_t)N_NODES * HEADS * 4);
  unsigned* mkey = (unsigned*)carve((size_t)N_NODES * HEADS * 4);
  float* ssum = (float*)carve((size_t)N_NODES * HEADS * 4);
  float* ebuf = (float*)carve((size_t)E_TOT * HEADS * 4);

  auto blocks = [](long n, int bs) { return (unsigned)((n + bs - 1) / bs); };

  // ---- stage 0: conversions ----
  k_f32_to_bf16<<<blocks((long)N_NODES * IN_DIM, 256), 256, 0, stream>>>(
      x, x_bf, (long)N_NODES * IN_DIM);
  k_transpose_bf16<<<blocks((long)V_DIM * HID, 256), 256, 0, stream>>>(
      vis_proj_w, wt_vproj, V_DIM, HID);
  k_transpose_bf16<<<blocks((long)T_DIM * HID, 256), 256, 0, stream>>>(
      txt_proj_w, wt_tproj, T_DIM, HID);
  for (int l = 0; l < LAYERS; ++l) {
    k_transpose_bf16<<<blocks((long)HID * HEADS * HID, 256), 256, 0, stream>>>(
        vis_gat_w + (size_t)l * HID * HEADS * HID, wt_gat[l], HID, HEADS * HID);
    k_transpose_bf16<<<blocks((long)HID * HEADS * HID, 256), 256, 0, stream>>>(
        txt_gat_w + (size_t)l * HID * HEADS * HID, wt_gat[2 + l], HID, HEADS * HID);
  }
  k_transpose_bf16<<<blocks((long)HID * HID, 256), 256, 0, stream>>>(
      vhead_w, wt_vhead, HID, HID);
  k_transpose_bf16<<<blocks((long)HID * HID, 256), 256, 0, stream>>>(
      thead_w, wt_thead, HID, HID);

  // ---- stage 1: input projections (WMMA, fused bias+ReLU) ----
  // N=128 -> 4 waves (128 threads) cover all 4 column tiles of 32
  dim3 gN128(N_NODES / 32, 1);
  k_gemm_wmma<<<gN128, 128, 0, stream>>>(x_bf, IN_DIM, wt_vproj, vis_proj_b,
                                         h_v, HID, HID, V_DIM, /*relu*/1);
  k_gemm_wmma<<<gN128, 128, 0, stream>>>(x_bf + V_DIM, IN_DIM, wt_tproj, txt_proj_b,
                                         h_t, HID, HID, T_DIM, /*relu*/1);

  // ---- stage 2: GAT layers ----
  auto run_gat = [&](float* hbuf, const unsigned short* gwt,
                     const float* gas, const float* gad, const float* gb) {
    // xh = h @ W  (20000x128 @ 128x512)
    k_f32_to_bf16<<<blocks((long)N_NODES * HID, 256), 256, 0, stream>>>(
        hbuf, h_bf, (long)N_NODES * HID);
    dim3 g2(N_NODES / 32, (HEADS * HID) / (32 * 8));   // 8 waves x 32 cols x grid.y=2
    k_gemm_wmma<<<g2, 256, 0, stream>>>(h_bf, HID, gwt, (const float*)nullptr,
                                        xh, HEADS * HID, HEADS * HID, HID, 0);
    // attention logits
    k_attn_logits<<<N_NODES, HEADS * 32, 0, stream>>>(xh, gas, gad, al_s, al_d);
    // init stats + agg
    k_fill_u32<<<blocks((long)N_NODES * HEADS, 256), 256, 0, stream>>>(
        mkey, 0u, (long)N_NODES * HEADS);
    k_fill_f32<<<blocks((long)N_NODES * HEADS, 256), 256, 0, stream>>>(
        ssum, 0.f, (long)N_NODES * HEADS);
    k_fill_f32<<<blocks((long)N_NODES * HEADS * HID, 256), 256, 0, stream>>>(
        agg, 0.f, (long)N_NODES * HEADS * HID);
    // segment softmax + aggregate
    k_edge_max<<<blocks((long)E_TOT * HEADS, 256), 256, 0, stream>>>(
        src, dst, al_s, al_d, mkey);
    k_edge_expsum<<<blocks((long)E_TOT * HEADS, 256), 256, 0, stream>>>(
        src, dst, al_s, al_d, mkey, ebuf, ssum);
    k_edge_aggregate<<<E_TOT, HEADS * 32, 0, stream>>>(
        src, dst, ebuf, ssum, xh, agg);
    // mean over heads + bias + ELU + residual
    k_node_update<<<blocks((long)N_NODES * HID, 256), 256, 0, stream>>>(
        agg, gb, hbuf);
  };

  for (int l = 0; l < LAYERS; ++l) {
    run_gat(h_v, wt_gat[l],     vis_gat_as + (size_t)l * HEADS * HID,
            vis_gat_ad + (size_t)l * HEADS * HID, vis_gat_b + (size_t)l * HID);
    run_gat(h_t, wt_gat[2 + l], txt_gat_as + (size_t)l * HEADS * HID,
            txt_gat_ad + (size_t)l * HEADS * HID, txt_gat_b + (size_t)l * HID);
  }

  // ---- stage 3: outputs ----
  float* out_emb  = (float*)d_out;                         // [N, HID]
  float* out_vis  = out_emb + (size_t)N_NODES * HID;       // [N, HID]
  float* out_txt  = out_vis + (size_t)N_NODES * HID;       // [N, HID]

  k_modal_final<<<blocks((long)N_NODES, 8), 256, 0, stream>>>(
      h_v, h_t, modal_w, modal_b, out_emb);

  k_f32_to_bf16<<<blocks((long)N_NODES * HID, 256), 256, 0, stream>>>(
      h_v, h_bf, (long)N_NODES * HID);
  k_gemm_wmma<<<gN128, 128, 0, stream>>>(h_bf, HID, wt_vhead, vhead_b,
                                         out_vis, HID, HID, HID, /*relu*/1);
  k_f32_to_bf16<<<blocks((long)N_NODES * HID, 256), 256, 0, stream>>>(
      h_t, h_bf, (long)N_NODES * HID);
  k_gemm_wmma<<<gN128, 128, 0, stream>>>(h_bf, HID, wt_thead, thead_b,
                                         out_txt, HID, HID, HID, /*relu*/1);
}
